// FluxBitSingleBlock_14474039787745
// MI455X (gfx1250) — 8-bit matrix operand order repaired and checked by reading, compiled, not run on hardware
//
#include <hip/hip_runtime.h>
#include <cstdint>
#include <cstddef>

#define BB 4
#define TT 4096
#define HH 768
#define DD 768
#define II 3072
#define MM (BB * TT)
#define NCHUNK 64
#define CLEN (TT / NCHUNK)

typedef int v8i __attribute__((ext_vector_type(8)));

__device__ __forceinline__ float block_sum256(float v, float* red) {
  int t = threadIdx.x;
  red[t] = v; __syncthreads();
  for (int off = 128; off > 0; off >>= 1) {
    if (t < off) red[t] += red[t + off];
    __syncthreads();
  }
  float r = red[0]; __syncthreads();
  return r;
}
__device__ __forceinline__ float block_max256(float v, float* red) {
  int t = threadIdx.x;
  red[t] = v; __syncthreads();
  for (int off = 128; off > 0; off >>= 1) {
    if (t < off) red[t] = fmaxf(red[t], red[t + off]);
    __syncthreads();
  }
  float r = red[0]; __syncthreads();
  return r;
}

__device__ __forceinline__ float silu_f(float x) { return x / (1.f + expf(-x)); }
__device__ __forceinline__ float sigm_f(float x) { return 1.f / (1.f + expf(-x)); }

__global__ __launch_bounds__(256)
void wsum_partial_kernel(const float* __restrict__ w, int n, float* __restrict__ partial) {
  __shared__ float red[256];
  float s = 0.f;
  for (int i = blockIdx.x * 256 + threadIdx.x; i < n; i += 64 * 256)
    s += fabsf(w[i]);
  float tot = block_sum256(s, red);
  if (threadIdx.x == 0) partial[blockIdx.x] = tot;
}

__global__ void wfinalize_kernel(const float* __restrict__ partial, int n,
                                 float* __restrict__ outsc) {
  if (threadIdx.x == 0 && blockIdx.x == 0) {
    float s = 0.f;
    for (int i = 0; i < 64; ++i) s += partial[i];
    outsc[0] = fmaxf(s / (float)n, 1e-5f);
  }
}

__global__ __launch_bounds__(256)
void wquant_kernel(const float* __restrict__ w, const float* __restrict__ dq,
                   int8_t* __restrict__ wq, int n) {
  float inv = 1.f / dq[0];
  for (int i = blockIdx.x * blockDim.x + threadIdx.x; i < n;
       i += gridDim.x * blockDim.x) {
    float q = rintf(w[i] * inv);
    q = fminf(fmaxf(q, -1.f), 1.f);
    wq[i] = (int8_t)q;
  }
}

__global__ __launch_bounds__(256)
void emb_kernel(const float* __restrict__ c, const float* __restrict__ w_emb,
                const float* __restrict__ b_emb, float* __restrict__ emb) {
  int idx = blockIdx.x * blockDim.x + threadIdx.x;
  if (idx >= BB * 2 * HH) return;
  int b = idx / (2 * HH);
  int n = idx % (2 * HH);
  const float* cr = c + (size_t)b * HH;
  const float* wr = w_emb + (size_t)n * HH;
  float s = 0.f;
  for (int k = 0; k < HH; ++k) s += silu_f(cr[k]) * wr[k];
  emb[idx] = s + b_emb[n];
}

__global__ __launch_bounds__(256)
void adaln_norm_quant_kernel(const float* __restrict__ x, const float* __restrict__ emb,
                             int8_t* __restrict__ nxq, float* __restrict__ nxscale) {
  __shared__ float red[256];
  const int m = blockIdx.x;
  const int b = m / TT;
  const int t = threadIdx.x;
  const float* xr = x + (size_t)m * HH;
  const float* sc = emb + (size_t)b * (2 * HH);
  float v[3];
  float s = 0.f;
#pragma unroll
  for (int j = 0; j < 3; ++j) { v[j] = xr[t + j * 256]; s += v[j]; }
  float mu = block_sum256(s, red) * (1.f / HH);
  float ss = 0.f;
#pragma unroll
  for (int j = 0; j < 3; ++j) { float d = v[j] - mu; ss += d * d; }
  float var = block_sum256(ss, red) * (1.f / HH);
  float rinv = rsqrtf(var + 1e-6f);
  float nx[3]; float ss2 = 0.f;
#pragma unroll
  for (int j = 0; j < 3; ++j) {
    int h = t + j * 256;
    nx[j] = (v[j] - mu) * rinv * (1.f + sc[h]);
    ss2 += nx[j] * nx[j];
  }
  float ms = block_sum256(ss2, red) * (1.f / HH);
  float rr = rsqrtf(ms + 1e-6f);
  float mx = 0.f;
#pragma unroll
  for (int j = 0; j < 3; ++j) { nx[j] *= rr; mx = fmaxf(mx, fabsf(nx[j])); }
  float rowmax = block_max256(mx, red);
  float dq = fmaxf(rowmax, 1e-5f) / 127.f;
  float qs = 1.f / dq;
#pragma unroll
  for (int j = 0; j < 3; ++j) {
    float q = fminf(fmaxf(rintf(nx[j] * qs), -128.f), 127.f);
    nxq[(size_t)m * HH + t + j * 256] = (int8_t)q;
  }
  if (t == 0) nxscale[m] = dq;
}

__global__ __launch_bounds__(256)
void gemm_i8_kernel(const int8_t* __restrict__ Aq, const float* __restrict__ arow,
                    const int8_t* __restrict__ Bq, const float* __restrict__ wscale,
                    float* __restrict__ C, int N, int K) {
  __shared__ int4 lds_a4[2][512];
  __shared__ int4 lds_b4[2][256];
  const int tid  = threadIdx.x;
  const int lane = tid & 31;
  const int wave = tid >> 5;
  const int l    = lane & 15;
  const int g    = lane >> 4;
  const int m0b  = blockIdx.x * 128;
  const int n0b  = blockIdx.y * 64;

  const int ar0 = m0b + (tid >> 2);
  const int ao0 = (tid & 3) * 16;
  const int ar1 = m0b + ((tid + 256) >> 2);
  const int ao1 = ((tid + 256) & 3) * 16;
  const int br  = n0b + (tid >> 2);
  const int bo  = (tid & 3) * 16;

  const int steps = K >> 6;

  v8i acc[4];
#pragma unroll
  for (int j = 0; j < 4; ++j)
#pragma unroll
    for (int v = 0; v < 8; ++v) acc[j][v] = 0;

  int4 ra0 = *(const int4*)(Aq + (size_t)ar0 * K + ao0);
  int4 ra1 = *(const int4*)(Aq + (size_t)ar1 * K + ao1);
  int4 rb  = *(const int4*)(Bq + (size_t)br  * K + bo);
  lds_a4[0][tid] = ra0;
  lds_a4[0][tid + 256] = ra1;
  lds_b4[0][tid] = rb;
  if (steps > 1) {
    ra0 = *(const int4*)(Aq + (size_t)ar0 * K + 64 + ao0);
    ra1 = *(const int4*)(Aq + (size_t)ar1 * K + 64 + ao1);
    rb  = *(const int4*)(Bq + (size_t)br  * K + 64 + bo);
  }
  __syncthreads();

  for (int ks = 0; ks < steps; ++ks) {
    const int buf = ks & 1;
    const int* la = (const int*)lds_a4[buf];
    const int* lb = (const int*)lds_b4[buf];

    v8i a;
#pragma unroll
    for (int v = 0; v < 8; ++v) {
      int ko = ((v >> 1) << 4) + g * 8 + ((v & 1) << 2);
      a[v] = la[(wave * 16 + l) * 16 + (ko >> 2)];
    }
    v8i bf[4];
#pragma unroll
    for (int j = 0; j < 4; ++j)
#pragma unroll
      for (int v = 0; v < 8; ++v) {
        int ko = ((v >> 1) << 4) + g * 8 + ((v & 1) << 2);
        bf[j][v] = lb[(j * 16 + l) * 16 + (ko >> 2)];
      }
#pragma unroll
    for (int j = 0; j < 4; ++j)
      acc[j] = __builtin_amdgcn_wmma_i32_16x16x64_iu8(
          true, a, true, bf[j], acc[j], false, false);

    if (ks + 1 < steps) {
      const int nb = buf ^ 1;
      lds_a4[nb][tid] = ra0;
      lds_a4[nb][tid + 256] = ra1;
      lds_b4[nb][tid] = rb;
      if (ks + 2 < steps) {
        const int k2 = (ks + 2) << 6;
        ra0 = *(const int4*)(Aq + (size_t)ar0 * K + k2 + ao0);
        ra1 = *(const int4*)(Aq + (size_t)ar1 * K + k2 + ao1);
        rb  = *(const int4*)(Bq + (size_t)br  * K + k2 + bo);
      }
    }
    __syncthreads();
  }

  const float ws = wscale[0];
#pragma unroll
  for (int j = 0; j < 4; ++j) {
    int n = n0b + j * 16 + l;
#pragma unroll
    for (int v = 0; v < 8; ++v) {
      int m = m0b + wave * 16 + v + g * 8;
      C[(size_t)m * N + n] = (float)acc[j][v] * arow[m] * ws;
    }
  }
}

__global__ __launch_bounds__(256)
void glu_gemm_i8_kernel(const int8_t* __restrict__ Aq, const float* __restrict__ arow,
                        const int8_t* __restrict__ Bq, const float* __restrict__ wscale,
                        float* __restrict__ U, int Nhalf, int K) {
  __shared__ int4 lds_a4[2][512];
  __shared__ int4 lds_bg4[2][128];
  __shared__ int4 lds_by4[2][128];
  const int tid  = threadIdx.x;
  const int lane = tid & 31;
  const int wave = tid >> 5;
  const int l    = lane & 15;
  const int g    = lane >> 4;
  const int m0b  = blockIdx.x * 128;
  const int n0b  = blockIdx.y * 32;

  const int ar0 = m0b + (tid >> 2);
  const int ao0 = (tid & 3) * 16;
  const int ar1 = m0b + ((tid + 256) >> 2);
  const int ao1 = ((tid + 256) & 3) * 16;
  const int isG  = (tid < 128) ? 1 : 0;
  const int bu   = isG ? tid : tid - 128;
  const int brow = (isG ? 0 : Nhalf) + n0b + (bu >> 2);
  const int boff = (bu & 3) * 16;

  const int steps = K >> 6;

  v8i accg[2], accy[2];
#pragma unroll
  for (int j = 0; j < 2; ++j)
#pragma unroll
    for (int v = 0; v < 8; ++v) { accg[j][v] = 0; accy[j][v] = 0; }

  int4 ra0 = *(const int4*)(Aq + (size_t)ar0 * K + ao0);
  int4 ra1 = *(const int4*)(Aq + (size_t)ar1 * K + ao1);
  int4 rb  = *(const int4*)(Bq + (size_t)brow * K + boff);
  lds_a4[0][tid] = ra0;
  lds_a4[0][tid + 256] = ra1;
  if (isG) lds_bg4[0][bu] = rb; else lds_by4[0][bu] = rb;
  if (steps > 1) {
    ra0 = *(const int4*)(Aq + (size_t)ar0 * K + 64 + ao0);
    ra1 = *(const int4*)(Aq + (size_t)ar1 * K + 64 + ao1);
    rb  = *(const int4*)(Bq + (size_t)brow * K + 64 + boff);
  }
  __syncthreads();

  for (int ks = 0; ks < steps; ++ks) {
    const int buf = ks & 1;
    const int* la = (const int*)lds_a4[buf];
    const int* lg = (const int*)lds_bg4[buf];
    const int* ly = (const int*)lds_by4[buf];

    v8i a;
#pragma unroll
    for (int v = 0; v < 8; ++v) {
      int ko = ((v >> 1) << 4) + g * 8 + ((v & 1) << 2);
      a[v] = la[(wave * 16 + l) * 16 + (ko >> 2)];
    }
    v8i bgf[2], byf[2];
#pragma unroll
    for (int j = 0; j < 2; ++j)
#pragma unroll
      for (int v = 0; v < 8; ++v) {
        int ko = ((v >> 1) << 4) + g * 8 + ((v & 1) << 2);
        bgf[j][v] = lg[(j * 16 + l) * 16 + (ko >> 2)];
        byf[j][v] = ly[(j * 16 + l) * 16 + (ko >> 2)];
      }
#pragma unroll
    for (int j = 0; j < 2; ++j) {
      accg[j] = __builtin_amdgcn_wmma_i32_16x16x64_iu8(
          true, a, true, bgf[j], accg[j], false, false);
      accy[j] = __builtin_amdgcn_wmma_i32_16x16x64_iu8(
          true, a, true, byf[j], accy[j], false, false);
    }

    if (ks + 1 < steps) {
      const int nb = buf ^ 1;
      lds_a4[nb][tid] = ra0;
      lds_a4[nb][tid + 256] = ra1;
      if (isG) lds_bg4[nb][bu] = rb; else lds_by4[nb][bu] = rb;
      if (ks + 2 < steps) {
        const int k2 = (ks + 2) << 6;
        ra0 = *(const int4*)(Aq + (size_t)ar0 * K + k2 + ao0);
        ra1 = *(const int4*)(Aq + (size_t)ar1 * K + k2 + ao1);
        rb  = *(const int4*)(Bq + (size_t)brow * K + k2 + boff);
      }
    }
    __syncthreads();
  }

  const float ws = wscale[0];
#pragma unroll
  for (int j = 0; j < 2; ++j) {
    int n = n0b + j * 16 + l;
#pragma unroll
    for (int v = 0; v < 8; ++v) {
      int m = m0b + wave * 16 + v + g * 8;
      float s  = arow[m] * ws;
      float yg = (float)accg[j][v] * s;
      float yy = (float)accy[j][v] * s;
      U[(size_t)m * Nhalf + n] = silu_f(yg) * yy;
    }
  }
}

__global__ __launch_bounds__(256)
void gates_kernel(float* __restrict__ ibuf, float* __restrict__ fbuf, int n) {
  for (int i = blockIdx.x * blockDim.x + threadIdx.x; i < n;
       i += gridDim.x * blockDim.x) {
    float f = sigm_f(fbuf[i]);
    float si = silu_f(ibuf[i]);
    ibuf[i] = si * (1.f - f);
    fbuf[i] = f;
  }
}

__global__ __launch_bounds__(256)
void scan_pass1_kernel(const float* __restrict__ ibuf, const float* __restrict__ fbuf,
                       float* __restrict__ Pc, float* __restrict__ Sc) {
  int tid = blockIdx.x * blockDim.x + threadIdx.x;
  if (tid >= BB * NCHUNK * DD) return;
  int d = tid % DD;
  int c = (tid / DD) % NCHUNK;
  int b = tid / (DD * NCHUNK);
  size_t base = ((size_t)b * TT + (size_t)c * CLEN) * DD + d;
  float P = 1.f, S = 0.f;
  for (int t = 0; t < CLEN; ++t) {
    float fv = fbuf[base + (size_t)t * DD];
    float iv = ibuf[base + (size_t)t * DD];
    S = fv * S + iv;
    P *= fv;
  }
  Pc[tid] = P; Sc[tid] = S;
}

__global__ __launch_bounds__(256)
void scan_pass2_kernel(const float* __restrict__ Pc, const float* __restrict__ Sc,
                       float* __restrict__ carry) {
  int tid = blockIdx.x * blockDim.x + threadIdx.x;
  if (tid >= BB * DD) return;
  int d = tid % DD;
  int b = tid / DD;
  float h = 0.f;
  for (int c = 0; c < NCHUNK; ++c) {
    size_t idx = ((size_t)b * NCHUNK + c) * DD + d;
    carry[idx] = h;
    h = Pc[idx] * h + Sc[idx];
  }
}

__global__ __launch_bounds__(256)
void scan_pass3_kernel(const float* __restrict__ ibuf, const float* __restrict__ fbuf,
                       const float* __restrict__ carry, float* __restrict__ obuf) {
  int tid = blockIdx.x * blockDim.x + threadIdx.x;
  if (tid >= BB * NCHUNK * DD) return;
  int d = tid % DD;
  int c = (tid / DD) % NCHUNK;
  int b = tid / (DD * NCHUNK);
  size_t base = ((size_t)b * TT + (size_t)c * CLEN) * DD + d;
  float h = carry[tid];
  for (int t = 0; t < CLEN; ++t) {
    size_t idx = base + (size_t)t * DD;
    h = fbuf[idx] * h + ibuf[idx];
    obuf[idx] = h;
  }
}

__global__ __launch_bounds__(256)
void o_gate_quant_kernel(const float* __restrict__ obuf, const float* __restrict__ gbuf,
                         const float* __restrict__ w_gn,
                         int8_t* __restrict__ oq, float* __restrict__ oscale) {
  __shared__ float red[256];
  const int m = blockIdx.x;
  const int t = threadIdx.x;
  const float* orow = obuf + (size_t)m * DD;
  const float* grow = gbuf + (size_t)m * DD;
  float ov[3]; float ss = 0.f;
#pragma unroll
  for (int j = 0; j < 3; ++j) { ov[j] = orow[t + j * 256]; ss += ov[j] * ov[j]; }
  float r1 = rsqrtf(block_sum256(ss, red) * (1.f / DD) + 1e-6f);
  float on[3]; float ss2 = 0.f;
#pragma unroll
  for (int j = 0; j < 3; ++j) {
    int h = t + j * 256;
    float og = ov[j] * r1 * w_gn[h] * silu_f(grow[h]);
    on[j] = og; ss2 += og * og;
  }
  float r2 = rsqrtf(block_sum256(ss2, red) * (1.f / DD) + 1e-6f);
  float mx = 0.f;
#pragma unroll
  for (int j = 0; j < 3; ++j) { on[j] *= r2; mx = fmaxf(mx, fabsf(on[j])); }
  float rowmax = block_max256(mx, red);
  float dq = fmaxf(rowmax, 1e-5f) / 127.f;
  float qs = 1.f / dq;
#pragma unroll
  for (int j = 0; j < 3; ++j) {
    float q = fminf(fmaxf(rintf(on[j] * qs), -128.f), 127.f);
    oq[(size_t)m * DD + t + j * 256] = (int8_t)q;
  }
  if (t == 0) oscale[m] = dq;
}

__global__ __launch_bounds__(256)
void u_quant_kernel(const float* __restrict__ ubuf,
                    int8_t* __restrict__ uq, float* __restrict__ uscale) {
  __shared__ float red[256];
  const int m = blockIdx.x;
  const int t = threadIdx.x;
  const float* ur = ubuf + (size_t)m * II;
  float v[12]; float ss = 0.f;
#pragma unroll
  for (int j = 0; j < 12; ++j) { v[j] = ur[t + j * 256]; ss += v[j] * v[j]; }
  float r = rsqrtf(block_sum256(ss, red) * (1.f / II) + 1e-6f);
  float mx = 0.f;
#pragma unroll
  for (int j = 0; j < 12; ++j) { v[j] *= r; mx = fmaxf(mx, fabsf(v[j])); }
  float rowmax = block_max256(mx, red);
  float dq = fmaxf(rowmax, 1e-5f) / 127.f;
  float qs = 1.f / dq;
#pragma unroll
  for (int j = 0; j < 12; ++j) {
    float q = fminf(fmaxf(rintf(v[j] * qs), -128.f), 127.f);
    uq[(size_t)m * II + t + j * 256] = (int8_t)q;
  }
  if (t == 0) uscale[m] = dq;
}

__global__ __launch_bounds__(256)
void hs_quant_kernel(const float* __restrict__ attn, const float* __restrict__ mlp,
                     int8_t* __restrict__ hsq, float* __restrict__ hscale) {
  __shared__ float red[256];
  const int m = blockIdx.x;
  const int t = threadIdx.x;
  float v[6]; float ss = 0.f;
#pragma unroll
  for (int j = 0; j < 6; ++j) {
    int h = t + j * 256;
    float val = (h < HH) ? attn[(size_t)m * HH + h] : mlp[(size_t)m * HH + (h - HH)];
    v[j] = val; ss += val * val;
  }
  float r = rsqrtf(block_sum256(ss, red) * (1.f / (2 * HH)) + 1e-6f);
  float mx = 0.f;
#pragma unroll
  for (int j = 0; j < 6; ++j) { v[j] *= r; mx = fmaxf(mx, fabsf(v[j])); }
  float rowmax = block_max256(mx, red);
  float dq = fmaxf(rowmax, 1e-5f) / 127.f;
  float qs = 1.f / dq;
#pragma unroll
  for (int j = 0; j < 6; ++j) {
    float q = fminf(fmaxf(rintf(v[j] * qs), -128.f), 127.f);
    hsq[(size_t)m * (2 * HH) + t + j * 256] = (int8_t)q;
  }
  if (t == 0) hscale[m] = dq;
}

__global__ __launch_bounds__(256)
void final_kernel(const float* __restrict__ x, const float* __restrict__ emb,
                  const float* __restrict__ ptmp, const float* __restrict__ b_proj,
                  float* __restrict__ out) {
  for (int idx = blockIdx.x * blockDim.x + threadIdx.x; idx < MM * HH;
       idx += gridDim.x * blockDim.x) {
    int m = idx / HH;
    int h = idx - m * HH;
    int b = m / TT;
    float gate = emb[(size_t)b * (2 * HH) + HH + h];
    out[idx] = x[idx] + gate * (ptmp[idx] + b_proj[h]);
  }
}

extern "C" void kernel_launch(void* const* d_in, const int* in_sizes, int n_in,
                              void* d_out, int out_size, void* d_ws, size_t ws_size,
                              hipStream_t stream) {
  (void)in_sizes; (void)n_in; (void)out_size; (void)ws_size;
  const float* x      = (const float*)d_in[0];
  const float* c      = (const float*)d_in[1];
  const float* w_emb  = (const float*)d_in[2];
  const float* b_emb  = (const float*)d_in[3];
  const float* w_i    = (const float*)d_in[4];
  const float* w_f    = (const float*)d_in[5];
  const float* w_g    = (const float*)d_in[6];
  const float* w_gn   = (const float*)d_in[7];
  const float* w_o    = (const float*)d_in[8];
  const float* w_gate = (const float*)d_in[9];
  const float* w_down = (const float*)d_in[10];
  const float* w_proj = (const float*)d_in[11];
  const float* b_proj = (const float*)d_in[12];
  float* out = (float*)d_out;

  uint8_t* ws = (uint8_t*)d_ws;
  size_t off = 0;
  auto alloc = [&](size_t bytes) -> void* {
    void* p = ws + off;
    off += (bytes + 255) & ~(size_t)255;
    return p;
  };

  int8_t* wq_i    = (int8_t*)alloc((size_t)DD * HH);
  int8_t* wq_f    = (int8_t*)alloc((size_t)DD * HH);
  int8_t* wq_g    = (int8_t*)alloc((size_t)DD * HH);
  int8_t* wq_o    = (int8_t*)alloc((size_t)HH * DD);
  int8_t* wq_gate = (int8_t*)alloc((size_t)2 * II * HH);
  int8_t* wq_down = (int8_t*)alloc((size_t)HH * II);
  int8_t* wq_proj = (int8_t*)alloc((size_t)HH * 2 * HH);
  float*  wscale  = (float*)alloc(7 * sizeof(float));
  float*  wpart   = (float*)alloc(7 * 64 * sizeof(float));
  float*  embbuf  = (float*)alloc((size_t)BB * 2 * HH * sizeof(float));
  int8_t* nxq     = (int8_t*)alloc((size_t)MM * HH);
  float*  nxscale = (float*)alloc((size_t)MM * sizeof(float));
  float*  ibuf    = (float*)alloc((size_t)MM * DD * sizeof(float));
  float*  fbuf    = (float*)alloc((size_t)MM * DD * sizeof(float));
  float*  gbuf    = (float*)alloc((size_t)MM * DD * sizeof(float));
  float*  obuf    = (float*)alloc((size_t)MM * DD * sizeof(float));
  int8_t* oq      = (int8_t*)alloc((size_t)MM * DD);
  float*  oscale  = (float*)alloc((size_t)MM * sizeof(float));
  float*  ubuf    = (float*)alloc((size_t)MM * II * sizeof(float));
  int8_t* uq      = (int8_t*)alloc((size_t)MM * II);
  float*  uscale  = (float*)alloc((size_t)MM * sizeof(float));
  float*  attn    = (float*)alloc((size_t)MM * HH * sizeof(float));
  float*  mlp     = (float*)alloc((size_t)MM * HH * sizeof(float));
  int8_t* hsq     = (int8_t*)alloc((size_t)MM * 2 * HH);
  float*  hscale  = (float*)alloc((size_t)MM * sizeof(float));
  float*  Pc      = (float*)alloc((size_t)BB * NCHUNK * DD * sizeof(float));
  float*  Sc      = (float*)alloc((size_t)BB * NCHUNK * DD * sizeof(float));
  float*  carry   = (float*)alloc((size_t)BB * NCHUNK * DD * sizeof(float));
  float*  ptmp    = ibuf;

  const float* wsrc[7] = {w_i, w_f, w_g, w_o, w_gate, w_down, w_proj};
  int8_t* wdst[7] = {wq_i, wq_f, wq_g, wq_o, wq_gate, wq_down, wq_proj};
  int wn[7] = {DD * HH, DD * HH, DD * HH, HH * DD, 2 * II * HH, HH * II, HH * 2 * HH};
  for (int t = 0; t < 7; ++t) {
    wsum_partial_kernel<<<64, 256, 0, stream>>>(wsrc[t], wn[t], wpart + t * 64);
    wfinalize_kernel<<<1, 1, 0, stream>>>(wpart + t * 64, wn[t], wscale + t);
    wquant_kernel<<<(wn[t] + 1023) / 1024, 256, 0, stream>>>(wsrc[t], wscale + t,
                                                             wdst[t], wn[t]);
  }

  emb_kernel<<<(BB * 2 * HH + 255) / 256, 256, 0, stream>>>(c, w_emb, b_emb, embbuf);
  adaln_norm_quant_kernel<<<MM, 256, 0, stream>>>(x, embbuf, nxq, nxscale);

  dim3 g768(MM / 128, HH / 64);
  gemm_i8_kernel<<<g768, 256, 0, stream>>>(nxq, nxscale, wq_i, wscale + 0, ibuf, HH, HH);
  gemm_i8_kernel<<<g768, 256, 0, stream>>>(nxq, nxscale, wq_f, wscale + 1, fbuf, HH, HH);
  gemm_i8_kernel<<<g768, 256, 0, stream>>>(nxq, nxscale, wq_g, wscale + 2, gbuf, HH, HH);

  gates_kernel<<<(MM * DD + 1023) / 1024, 256, 0, stream>>>(ibuf, fbuf, MM * DD);
  scan_pass1_kernel<<<(BB * NCHUNK * DD + 255) / 256, 256, 0, stream>>>(ibuf, fbuf, Pc, Sc);
  scan_pass2_kernel<<<(BB * DD + 255) / 256, 256, 0, stream>>>(Pc, Sc, carry);
  scan_pass3_kernel<<<(BB * NCHUNK * DD + 255) / 256, 256, 0, stream>>>(ibuf, fbuf, carry, obuf);

  o_gate_quant_kernel<<<MM, 256, 0, stream>>>(obuf, gbuf, w_gn, oq, oscale);
  gemm_i8_kernel<<<g768, 256, 0, stream>>>(oq, oscale, wq_o, wscale + 3, attn, HH, DD);

  glu_gemm_i8_kernel<<<dim3(MM / 128, II / 32), 256, 0, stream>>>(
      nxq, nxscale, wq_gate, wscale + 4, ubuf, II, HH);
  u_quant_kernel<<<MM, 256, 0, stream>>>(ubuf, uq, uscale);
  gemm_i8_kernel<<<g768, 256, 0, stream>>>(uq, uscale, wq_down, wscale + 5, mlp, HH, II);

  hs_quant_kernel<<<MM, 256, 0, stream>>>(attn, mlp, hsq, hscale);
  gemm_i8_kernel<<<g768, 256, 0, stream>>>(hsq, hscale, wq_proj, wscale + 6, ptmp, HH, 2 * HH);
  final_kernel<<<(MM * HH + 1023) / 1024, 256, 0, stream>>>(x, embbuf, ptmp, b_proj, out);
}
